// SimpleCTRNN_50723563766282
// MI455X (gfx1250) — matrix operand order repaired and checked by reading, not compiled, not run on hardware
//
#include <hip/hip_runtime.h>
#include <hip/hip_bf16.h>

#define DEV __device__ __forceinline__

typedef __attribute__((ext_vector_type(16))) __bf16 v16bf;
typedef __attribute__((ext_vector_type(8)))  float  v8f;
typedef __attribute__((ext_vector_type(4)))  float  f32x4;
typedef unsigned short u16;
typedef __attribute__((ext_vector_type(8)))  u16 u16x8;
typedef __attribute__((ext_vector_type(16))) u16 u16x16;
typedef __attribute__((ext_vector_type(2)))  unsigned u32x2;

// Problem dims
constexpr int S  = 512;
constexpr int B  = 128;
constexpr int DI = 512;
constexpr int DH = 1024;
constexpr int DO = 512;
constexpr int SB = S * B;

// Workspace layout (bytes)
constexpr size_t WH_OFF   = 0;                                   // [64 nt][32 kt][512] u16 = 2 MB
constexpr size_t WIN_OFF  = 2u * 1024 * 1024;                    // [64 nt][16 kt][512] u16 = 1 MB
constexpr size_t WOUT_OFF = 3u * 1024 * 1024;                    // [32 nt][32 kt][512] u16 = 1 MB
constexpr size_t HBF_OFF  = 4u * 1024 * 1024;                    // double-buffered h: 2*[B][DH] u16
constexpr size_t BAR_OFF  = HBF_OFF + (size_t)2 * B * DH * 2;    // 2 uints (count, generation)
constexpr size_t XBF_OFF  = 8u * 1024 * 1024;                    // x in bf16: [SB][DI] = 67 MB
constexpr size_t RBF_OFF  = XBF_OFF + (size_t)SB * DI * 2;       // rnn trace bf16: [S][B][DH] = 134 MB

DEV u16 f2bf(float f) {
  unsigned u = __builtin_bit_cast(unsigned, f);
  unsigned rnd = 0x7FFFu + ((u >> 16) & 1u);   // round-to-nearest-even
  return (u16)((u + rnd) >> 16);
}

DEV unsigned cvt2(float a, float b) {
  return (unsigned)f2bf(a) | ((unsigned)f2bf(b) << 16);
}

DEV v8f wmma_bf16(u16x16 a, u16x16 b, v8f c) {
  return __builtin_amdgcn_wmma_f32_16x16x32_bf16(
      /*neg_a=*/false, __builtin_bit_cast(v16bf, a),
      /*neg_b=*/false, __builtin_bit_cast(v16bf, b),
      /*c_mod=*/(short)0, c, /*reuse_a=*/false, /*reuse_b=*/false);
}

// Load a 16x32 A-fragment (contiguous k = half*16 + e, same order as pack_w) from a row-major bf16 row pointer.
DEV u16x16 ld_afrag(const u16* arow, int kt, int half) {
  u16x16 af = *(const u16x16*)(arow + kt * 32 + half * 16);     // K = half*16 + [0..15]
  return af;
}

// ---------------------------------------------------------------------------
// Pack weight W[N][K] (row-major, B[k][n] = W[n][k]) into WMMA B-fragment
// order: per 32x16 (KxN) tile, 512 bf16 where element (lane*16 + e) holds
// W[nt*16 + (lane&15)][kt*32 + (lane>>4)*16 + e].  Tiles stored kt-minor.
// ---------------------------------------------------------------------------
__global__ void __launch_bounds__(256) pack_w(const float* __restrict__ W,
                                              u16* __restrict__ dst,
                                              int N, int K) {
  int idx = blockIdx.x * blockDim.x + threadIdx.x;
  if (idx >= N * K) return;
  int tile   = idx >> 9;
  int within = idx & 511;
  int lane   = within >> 4;
  int e      = within & 15;
  int ktiles = K >> 5;
  int nt = tile / ktiles;
  int kt = tile % ktiles;
  int n  = nt * 16 + (lane & 15);
  int k  = kt * 32 + (lane >> 4) * 16 + e;
  dst[idx] = f2bf(W[(size_t)n * K + k]);
}

// Elementwise f32 -> bf16 (4 elements / thread, vectorized)
__global__ void __launch_bounds__(256) cvt_f32_bf16(const float* __restrict__ src,
                                                    u16* __restrict__ dst, int n4) {
  int i = blockIdx.x * blockDim.x + threadIdx.x;
  if (i >= n4) return;
  f32x4 v = ((const f32x4*)src)[i];
  u32x2 r;
  r[0] = cvt2(v[0], v[1]);
  r[1] = cvt2(v[2], v[3]);
  ((u32x2*)dst)[i] = r;
}

__global__ void __launch_bounds__(256) init_state(u16* __restrict__ hbf,
                                                  unsigned* __restrict__ bar,
                                                  int n) {
  int idx = blockIdx.x * blockDim.x + threadIdx.x;
  if (idx < n) hbf[idx] = 0;
  if (idx < 2) bar[idx] = 0u;
}

// ---------------------------------------------------------------------------
// Tiled GEMM:  C[M][N] = A[M][K](bf16 row-major) * Bpacked + bias
// Block = 8 waves sharing one N-group (4 N-tiles) with different M-tiles.
// B tiles are staged through double-buffered LDS (8 KB), so each 4 KB of
// B is fetched from L2 once per block instead of once per wave (8x less).
// Per K-step per wave: 2 global A loads + 4 LDS B reads + 4 WMMAs.
// ---------------------------------------------------------------------------
__global__ void __launch_bounds__(256) gemm_bf16(const u16* __restrict__ A,
                                                 const u16* __restrict__ Bp,
                                                 const float* __restrict__ bias,
                                                 float* __restrict__ C,
                                                 int Mtiles, int Ngroups, int Ktiles) {
  __shared__ __align__(16) u16 sB[2][4 * 512];          // 2 x 4 tiles x 1 KB

  int widx = threadIdx.x >> 5;                          // 0..7
  int bm   = blockIdx.x / Ngroups;                      // M-block (8 M-tiles)
  int tg   = blockIdx.x % Ngroups;                      // N-group
  int tm   = bm * 8 + widx;
  int tn0  = tg * 4;
  int lane = threadIdx.x & 31;
  int half = lane >> 4;
  int lr   = lane & 15;
  int K    = Ktiles * 32;

  const u16* arow = A + (size_t)(tm * 16 + lr) * K;

  // staging assignment: thread stages 16 B (8 u16) of tile sq at offset so
  int tid = threadIdx.x;
  int sq  = tid >> 6;                                   // 0..3 (tile)
  int so  = (tid & 63) * 8;                             // 0..504
  const u16* gstage = Bp + ((size_t)(tn0 + sq) * Ktiles) * 512 + so;

  v8f acc0 = {0.f, 0.f, 0.f, 0.f, 0.f, 0.f, 0.f, 0.f};
  v8f acc1 = acc0, acc2 = acc0, acc3 = acc0;

  // preload kt=0
  {
    u16x8 v = *(const u16x8*)(gstage);
    *(u16x8*)&sB[0][sq * 512 + so] = v;
  }
  __syncthreads();

  for (int kt = 0; kt < Ktiles; ++kt) {
    int cur = kt & 1;
    bool have_next = (kt + 1) < Ktiles;
    u16x8 nv;
    if (have_next) {
      nv = *(const u16x8*)(gstage + (size_t)(kt + 1) * 512);   // global fetch of next B
      __builtin_prefetch(arow + (kt + 1) * 32, 0, 1);
    }

    u16x16 af = ld_afrag(arow, kt, half);
    u16x16 b0 = *(const u16x16*)&sB[cur][0 * 512 + lane * 16];
    u16x16 b1 = *(const u16x16*)&sB[cur][1 * 512 + lane * 16];
    u16x16 b2 = *(const u16x16*)&sB[cur][2 * 512 + lane * 16];
    u16x16 b3 = *(const u16x16*)&sB[cur][3 * 512 + lane * 16];

    acc0 = wmma_bf16(af, b0, acc0);
    acc1 = wmma_bf16(af, b1, acc1);
    acc2 = wmma_bf16(af, b2, acc2);
    acc3 = wmma_bf16(af, b3, acc3);

    if (have_next) {
      *(u16x8*)&sB[cur ^ 1][sq * 512 + so] = nv;        // stage next buffer
    }
    __syncthreads();
  }

  int ldc = Ngroups * 64;
#pragma unroll
  for (int q = 0; q < 4; ++q) {
    v8f acc = q == 0 ? acc0 : q == 1 ? acc1 : q == 2 ? acc2 : acc3;
    int col = (tn0 + q) * 16 + lr;
    float bia = bias[col];
#pragma unroll
    for (int r = 0; r < 8; ++r) {
      int m = tm * 16 + r + half * 8;                   // C layout: row = r + 8*half
      C[(size_t)m * ldc + col] = acc[r] + bia;
    }
  }
}

// ---------------------------------------------------------------------------
// Persistent recurrent scan: 64 WGs (block = one N-tile tn, wave = one
// M-tile) x 8 waves.  The block's entire W_h column strip (32 KB, contiguous
// in packed layout) is preloaded into LDS once; all 512 timesteps read B
// from LDS only.  Per step: 32 chained bf16 WMMAs (K=1024), epilogue adds
// xp (resident in rnn_out region of d_out) + bias, relu, writes f32
// rnn_out, bf16 h double buffer and bf16 trace, then device-wide barrier.
// ---------------------------------------------------------------------------
__global__ void __launch_bounds__(256) scan_rnn(const u16* __restrict__ Whp,
                                                const float* __restrict__ b_h,
                                                float* __restrict__ rnn,   // [S][B][DH], holds xp on entry
                                                u16* __restrict__ hbf,     // 2*[B][DH]
                                                u16* __restrict__ rbf,     // [S][B][DH] bf16 trace
                                                unsigned* __restrict__ bar) {
  __shared__ __align__(16) u16 sW[32 * 512];            // 32 K-tiles x 1 KB = 32 KB

  int tn = blockIdx.x;                                  // 0..63 (DH/16)
  int tm = threadIdx.x >> 5;                            // 0..7  (B/16)
  int lane = threadIdx.x & 31;
  int half = lane >> 4;
  int lr   = lane & 15;
  int col  = tn * 16 + lr;
  float bias = b_h[col];

  // Preload the full W_h strip for this tn (contiguous 32 KB)
  {
    const u16x8* src = (const u16x8*)(Whp + (size_t)tn * 32 * 512);
    u16x8* dst = (u16x8*)sW;
    for (int i = threadIdx.x; i < (32 * 512) / 8; i += 256) dst[i] = src[i];
  }
  __syncthreads();

  unsigned* cnt = bar;
  unsigned* gen = bar + 1;
  const unsigned NWG = gridDim.x;

  for (int t = 0; t < S; ++t) {
    const u16* arow = hbf + (size_t)(t & 1) * B * DH + (size_t)(tm * 16 + lr) * DH;
    u16* wbuf = hbf + (size_t)((t + 1) & 1) * B * DH;

    v8f acc = {0.f, 0.f, 0.f, 0.f, 0.f, 0.f, 0.f, 0.f};
#pragma unroll 8
    for (int kt = 0; kt < DH / 32; ++kt) {
      u16x16 af = ld_afrag(arow, kt, half);
      u16x16 bfv = *(const u16x16*)&sW[kt * 512 + lane * 16];
      acc = wmma_bf16(af, bfv, acc);
    }

    float* rbase = rnn + (size_t)t * B * DH;
    u16*   tbase = rbf + (size_t)t * B * DH;
#pragma unroll
    for (int r = 0; r < 8; ++r) {
      int m = tm * 16 + r + half * 8;
      size_t off = (size_t)m * DH + col;
      float v = acc[r] + rbase[off] + bias;             // + xp_t + b_h
      v = v > 0.f ? v : 0.f;                            // relu; alpha==1 -> h = h_new
      u16 hv = f2bf(v);
      rbase[off] = v;                                   // rnn_out[t] (f32 output)
      wbuf[off] = hv;                                   // next-step h (bf16)
      tbase[off] = hv;                                  // bf16 trace for phase 3
    }

    // ---- device-wide barrier between timesteps ----
    __threadfence();
    __syncthreads();
    if (threadIdx.x == 0) {
      unsigned g = __hip_atomic_load(gen, __ATOMIC_ACQUIRE, __HIP_MEMORY_SCOPE_AGENT);
      unsigned old = __hip_atomic_fetch_add(cnt, 1u, __ATOMIC_ACQ_REL, __HIP_MEMORY_SCOPE_AGENT);
      if (old == NWG - 1) {
        __hip_atomic_store(cnt, 0u, __ATOMIC_RELAXED, __HIP_MEMORY_SCOPE_AGENT);
        __hip_atomic_fetch_add(gen, 1u, __ATOMIC_ACQ_REL, __HIP_MEMORY_SCOPE_AGENT);
      } else {
        while (__hip_atomic_load(gen, __ATOMIC_ACQUIRE, __HIP_MEMORY_SCOPE_AGENT) == g) {
          __builtin_amdgcn_s_sleep(1);
        }
      }
    }
    __syncthreads();
  }
}

extern "C" void kernel_launch(void* const* d_in, const int* in_sizes, int n_in,
                              void* d_out, int out_size, void* d_ws, size_t ws_size,
                              hipStream_t stream) {
  (void)in_sizes; (void)n_in; (void)out_size; (void)ws_size;

  const float* x     = (const float*)d_in[0];   // [S][B][DI]
  const float* W_in  = (const float*)d_in[1];   // [DH][DI]
  const float* b_in  = (const float*)d_in[2];   // [DH]
  const float* W_h   = (const float*)d_in[3];   // [DH][DH]
  const float* b_h   = (const float*)d_in[4];   // [DH]
  const float* W_out = (const float*)d_in[5];   // [DO][DH]
  const float* b_out = (const float*)d_in[6];   // [DO]

  float* out = (float*)d_out;                   // [S][B][DO]
  float* rnn = out + (size_t)SB * DO;           // [S][B][DH] (xp, then rnn_out)

  char* ws = (char*)d_ws;
  u16* whp   = (u16*)(ws + WH_OFF);
  u16* winp  = (u16*)(ws + WIN_OFF);
  u16* woutp = (u16*)(ws + WOUT_OFF);
  u16* hbf   = (u16*)(ws + HBF_OFF);
  unsigned* bar = (unsigned*)(ws + BAR_OFF);
  u16* xbf   = (u16*)(ws + XBF_OFF);
  u16* rbf   = (u16*)(ws + RBF_OFF);

  // Phase 0: weight packing, x -> bf16, state init
  pack_w<<<(DH * DH + 255) / 256, 256, 0, stream>>>(W_h,   whp,   DH, DH);
  pack_w<<<(DH * DI + 255) / 256, 256, 0, stream>>>(W_in,  winp,  DH, DI);
  pack_w<<<(DO * DH + 255) / 256, 256, 0, stream>>>(W_out, woutp, DO, DH);
  cvt_f32_bf16<<<(SB * DI / 4 + 255) / 256, 256, 0, stream>>>(x, xbf, SB * DI / 4);
  init_state<<<(2 * B * DH + 255) / 256, 256, 0, stream>>>(hbf, bar, 2 * B * DH);

  // Phase 1: xp = x @ W_in^T + b_in   (written into rnn_out region of d_out)
  {
    int mt = SB / 16, ng = (DH / 16) / 4, kt = DI / 32;   // 4096, 16, 16
    gemm_bf16<<<(mt / 8) * ng, 256, 0, stream>>>(xbf, winp, b_in, rnn, mt, ng, kt);
  }

  // Phase 2: persistent recurrent scan (512 steps, in-kernel grid barrier)
  scan_rnn<<<64, 256, 0, stream>>>(whp, b_h, rnn, hbf, rbf, bar);

  // Phase 3: out = rnn_out @ W_out^T + b_out
  {
    int mt = SB / 16, ng = (DO / 16) / 4, kt = DH / 32;   // 4096, 8, 32
    gemm_bf16<<<(mt / 8) * ng, 256, 0, stream>>>(rbf, woutp, b_out, out, mt, ng, kt);
  }
}
